// SimpleBEV_7378753814752
// MI455X (gfx1250) — compile-verified, matrix operand order checked by reading, not run on hardware
//
#include <hip/hip_runtime.h>
#include <math.h>

#define ZDIM 128
#define YDIM 8
#define XDIM 128
#define CDIM 128
#define ICDIM 1024            // C*Y
#define HF 56
#define WF 100
#define SCAMS 6
#define SPATIAL (ZDIM * XDIM) // 16384
#define XP 130                // padded spatial extent (halo of 1 on each side)
#define BEV_ELEMS ((size_t)XP * XP * ICDIM)   // padded bev, bf16 elements

typedef __attribute__((ext_vector_type(16))) __bf16 v16bf;
typedef __attribute__((ext_vector_type(8)))  float  v8f;

union Frag16 { v16bf v; uint4 u[2]; };

__device__ __forceinline__ unsigned short f2bf(float f) {
    union { float f; unsigned u; } a;
    a.f = f;
    unsigned u = a.u;
    unsigned r = (u + 0x7FFFu + ((u >> 16) & 1u)) >> 16; // RNE
    return (unsigned short)r;
}

// ---------------------------------------------------------------------------
// Kernel 0: per-camera transforms (analytic rigid inverse; no 4x4 inversion)
//   xf[0   + s*16 .. ]  = camXs_T_cam0[s]   (row-major 4x4)
//   xf[96  + s*16 .. ]  = pixB_T_cam0[s]
// ---------------------------------------------------------------------------
__global__ void xform_kernel(const float* __restrict__ intrins,
                             const float* __restrict__ rots,
                             const float* __restrict__ trans,
                             float* __restrict__ xf) {
    int s = threadIdx.x;
    if (s >= SCAMS) return;
    float Rs[9], R0[9], ts[3], t0[3];
    #pragma unroll
    for (int i = 0; i < 9; i++) { Rs[i] = rots[s * 9 + i]; R0[i] = rots[i]; }
    #pragma unroll
    for (int i = 0; i < 3; i++) { ts[i] = trans[s * 3 + i]; t0[i] = trans[i]; }
    // camXs_T_cam0 = inv([Rs|ts]) @ [R0|t0] = [Rs^T R0 | Rs^T (t0 - ts)]
    float M[16];
    #pragma unroll
    for (int i = 0; i < 3; i++) {
        #pragma unroll
        for (int j = 0; j < 3; j++) {
            float acc = 0.f;
            #pragma unroll
            for (int k = 0; k < 3; k++) acc += Rs[k * 3 + i] * R0[k * 3 + j];
            M[i * 4 + j] = acc;
        }
        float acc = 0.f;
        #pragma unroll
        for (int k = 0; k < 3; k++) acc += Rs[k * 3 + i] * (t0[k] - ts[k]);
        M[i * 4 + 3] = acc;
    }
    M[12] = 0.f; M[13] = 0.f; M[14] = 0.f; M[15] = 1.f;

    const float sx = (float)WF / 800.f, sy = (float)HF / 448.f;
    float fx = intrins[s * 16 + 0] * sx, fy = intrins[s * 16 + 5] * sy;
    float cx = intrins[s * 16 + 2] * sx, cy = intrins[s * 16 + 6] * sy;
    float P[16];
    #pragma unroll
    for (int j = 0; j < 4; j++) {
        P[0 * 4 + j] = fx * M[0 * 4 + j] + cx * M[2 * 4 + j];
        P[1 * 4 + j] = fy * M[1 * 4 + j] + cy * M[2 * 4 + j];
        P[2 * 4 + j] = M[2 * 4 + j];
        P[3 * 4 + j] = (j == 3) ? 1.f : 0.f;
    }
    #pragma unroll
    for (int i = 0; i < 16; i++) { xf[s * 16 + i] = M[i]; xf[96 + s * 16 + i] = P[i]; }
}

// ---------------------------------------------------------------------------
// Kernel 1a: zero padded BEV buffer (uint4 = 8 bf16 per store)
// ---------------------------------------------------------------------------
__global__ void bev_zero_kernel(uint4* __restrict__ p, long n4) {
    long i = (long)blockIdx.x * blockDim.x + threadIdx.x;
    if (i < n4) p[i] = make_uint4(0u, 0u, 0u, 0u);
}

// ---------------------------------------------------------------------------
// Kernel 1b: repack conv weights (oc,ic,3,3) f32 -> [tap][oc][ic] bf16
// ---------------------------------------------------------------------------
__global__ void wpack_kernel(const float* __restrict__ w,
                             unsigned short* __restrict__ wp, int n) {
    int i = blockIdx.x * blockDim.x + threadIdx.x;
    if (i >= n) return;
    int ic  = i & (ICDIM - 1);
    int oc  = (i / ICDIM) & (CDIM - 1);
    int tap = i / (ICDIM * CDIM);
    int ky = tap / 3, kx = tap % 3;
    wp[i] = f2bf(w[((size_t)(oc * ICDIM + ic) * 3 + ky) * 3 + kx]);
}

// ---------------------------------------------------------------------------
// Kernel 2: project voxels, bilinear-sample all 6 cameras, masked camera mean,
// write bf16 BEV tensor in padded [z+1][x+1][ic=c*8+y] layout (WMMA-ready,
// K-contiguous, zero halo). One wave per voxel; lanes stride the 128 channels.
// ---------------------------------------------------------------------------
__global__ void sample_kernel(const float* __restrict__ cam_feat,
                              const float* __restrict__ xf,
                              unsigned short* __restrict__ bev) {
    const int wave = threadIdx.x >> 5;
    const int lane = threadIdx.x & 31;
    const int v = blockIdx.x * 8 + wave;     // 0 .. 131071
    const int z = v >> 10;
    const int y = (v >> 7) & 7;
    const int x = v & 127;

    const float vs = 100.f / 128.f;
    const float xr = -50.f + 0.5f * vs + x * vs;
    const float zr = -50.f + 0.5f * vs + z * vs;
    const float yr = -4.9f + 0.625f + y * 1.25f;

    float acc[4] = {0.f, 0.f, 0.f, 0.f};
    float cnt[4] = {0.f, 0.f, 0.f, 0.f};

    for (int s = 0; s < SCAMS; s++) {
        const float* M = xf + s * 16;
        const float* P = xf + 96 + s * 16;
        float camz = M[8] * xr + M[9] * yr + M[10] * zr + M[11];
        float px4  = P[0] * xr + P[1] * yr + P[2]  * zr + P[3];
        float py4  = P[4] * xr + P[5] * yr + P[6]  * zr + P[7];
        float pz4  = P[8] * xr + P[9] * yr + P[10] * zr + P[11];
        float denom = fmaxf(pz4, 1e-6f);
        float px = px4 / denom, py = py4 / denom;
        bool valid = (px > -0.5f) && (px < WF - 0.5f) &&
                     (py > -0.5f) && (py < HF - 0.5f) && (camz > 0.f);
        if (!valid) continue;   // uniform across the wave (same voxel)

        float gx = fminf(fmaxf(2.f * px / (WF - 1) - 1.f, -2.f), 2.f);
        float gy = fminf(fmaxf(2.f * py / (HF - 1) - 1.f, -2.f), 2.f);
        float ux = ((gx + 1.f) * WF - 1.f) * 0.5f;
        float uy = ((gy + 1.f) * HF - 1.f) * 0.5f;
        float x0f = floorf(ux), y0f = floorf(uy);
        float wx = ux - x0f, wy = uy - y0f;
        int x0 = (int)x0f, y0 = (int)y0f, x1 = x0 + 1, y1 = y0 + 1;
        float w00 = (1.f - wx) * (1.f - wy), w10 = wx * (1.f - wy);
        float w01 = (1.f - wx) * wy,         w11 = wx * wy;
        bool bx0 = (x0 >= 0) && (x0 < WF), bx1 = (x1 >= 0) && (x1 < WF);
        bool by0 = (y0 >= 0) && (y0 < HF), by1 = (y1 >= 0) && (y1 < HF);
        w00 = (bx0 && by0) ? w00 : 0.f;  w10 = (bx1 && by0) ? w10 : 0.f;
        w01 = (bx0 && by1) ? w01 : 0.f;  w11 = (bx1 && by1) ? w11 : 0.f;
        int cx0 = min(max(x0, 0), WF - 1), cx1 = min(max(x1, 0), WF - 1);
        int cy0 = min(max(y0, 0), HF - 1), cy1 = min(max(y1, 0), HF - 1);
        int i00 = cy0 * WF + cx0, i10 = cy0 * WF + cx1;
        int i01 = cy1 * WF + cx0, i11 = cy1 * WF + cx1;
        const float* base = cam_feat + (size_t)s * CDIM * HF * WF;
        #pragma unroll
        for (int j = 0; j < 4; j++) {
            const float* f = base + (size_t)(lane + 32 * j) * (HF * WF);
            float val = w00 * f[i00] + w10 * f[i10] + w01 * f[i01] + w11 * f[i11];
            acc[j] += val;
            cnt[j] += (val != 0.f) ? 1.f : 0.f;
        }
    }
    unsigned short* ob = bev + ((size_t)((z + 1) * XP + (x + 1))) * ICDIM + y;
    #pragma unroll
    for (int j = 0; j < 4; j++) {
        int c = lane + 32 * j;
        ob[c * YDIM] = f2bf(acc[j] / (1e-6f + cnt[j]));
    }
}

// ---------------------------------------------------------------------------
// Kernel 3: 3x3 conv as implicit GEMM via v_wmma_f32_16x16x32_bf16.
// GEMM: M = oc (128), N = x (tile 64, fixed z), K = ic (1024) x 9 taps.
// Zero-padded input -> completely branch-free inner loop, EXEC constant.
// Block = 4 waves (128 thr), tile 128(oc) x 64(x). Each wave owns
// 4 M-subtiles x 2 N-subtiles = 8 accumulators: per (tap,k-step) it loads
// 4 A frags + 2 B frags (12 x b128) and issues 8 WMMAs (1.5 loads/WMMA).
// ---------------------------------------------------------------------------
__global__ void conv_kernel(const unsigned short* __restrict__ bev,
                            const unsigned short* __restrict__ wp,
                            float* __restrict__ out) {
    const int lane    = threadIdx.x & 31;
    const int wave    = threadIdx.x >> 5;   // 0..3
    const int lane_lo = lane & 15;
    const int khalf   = (lane >> 4) << 3;   // 0 or 8 (ISA 16-bit frag K split)
    const int m_half  = wave & 1;           // 64 oc each
    const int n_pair  = wave >> 1;          // 32 x each
    const int z  = blockIdx.y;
    const int x0 = blockIdx.x * 64 + n_pair * 32;

    const v8f zero8 = {0.f, 0.f, 0.f, 0.f, 0.f, 0.f, 0.f, 0.f};
    v8f acc[8] = {zero8, zero8, zero8, zero8, zero8, zero8, zero8, zero8};

    #pragma unroll
    for (int ky = 0; ky < 3; ky++) {
        #pragma unroll
        for (int kx = 0; kx < 3; kx++) {
            const int tap = ky * 3 + kx;
            // padded column for B-frag lane: global x = x0 + n + lane_lo + kx - 1
            const unsigned short* brow =
                bev + ((size_t)((z + ky) * XP + (x0 + lane_lo + kx))) * ICDIM + khalf;
            const unsigned short* wrow =
                wp + ((size_t)(tap * CDIM + m_half * 64 + lane_lo)) * ICDIM + khalf;
            for (int kb = 0; kb < 32; kb++) {
                const int k0 = kb * 32;
                Frag16 Bf[2];
                #pragma unroll
                for (int j = 0; j < 2; j++) {
                    const unsigned short* p = brow + (size_t)j * 16 * ICDIM + k0;
                    Bf[j].u[0] = *(const uint4*)(p);
                    Bf[j].u[1] = *(const uint4*)(p + 16);
                }
                Frag16 Af[4];
                #pragma unroll
                for (int i = 0; i < 4; i++) {
                    const unsigned short* p = wrow + (size_t)i * 16 * ICDIM + k0;
                    Af[i].u[0] = *(const uint4*)(p);
                    Af[i].u[1] = *(const uint4*)(p + 16);
                }
                #pragma unroll
                for (int i = 0; i < 4; i++)
                    #pragma unroll
                    for (int j = 0; j < 2; j++)
                        acc[i * 2 + j] = __builtin_amdgcn_wmma_f32_16x16x32_bf16(
                            false, Af[i].v, false, Bf[j].v, (short)0,
                            acc[i * 2 + j], false, false);
            }
        }
    }
    // D layout: VGPR r, lanes 0-15 -> M=r, lanes 16-31 -> M=8+r; N = lane&15
    const int mhalf = (lane >> 4) * 8;
    #pragma unroll
    for (int i = 0; i < 4; i++) {
        const int ocb = m_half * 64 + i * 16 + mhalf;
        #pragma unroll
        for (int j = 0; j < 2; j++) {
            const int xo = x0 + j * 16 + lane_lo;
            #pragma unroll
            for (int r = 0; r < 8; r++) {
                out[(size_t)(ocb + r) * SPATIAL + z * XDIM + xo] = acc[i * 2 + j][r];
            }
        }
    }
}

// ---------------------------------------------------------------------------
// Kernel 4: InstanceNorm (per output channel over 128x128) + exact GELU
// ---------------------------------------------------------------------------
__global__ void norm_gelu_kernel(const float* __restrict__ cin,
                                 float* __restrict__ outp) {
    const int oc = blockIdx.x;
    const float* src = cin + (size_t)oc * SPATIAL;
    __shared__ float s_sum[256];
    __shared__ float s_sq[256];
    float sum = 0.f, sq = 0.f;
    for (int i = threadIdx.x; i < SPATIAL; i += 256) {
        float v = src[i];
        sum += v; sq += v * v;
    }
    s_sum[threadIdx.x] = sum; s_sq[threadIdx.x] = sq;
    __syncthreads();
    for (int off = 128; off > 0; off >>= 1) {
        if (threadIdx.x < off) {
            s_sum[threadIdx.x] += s_sum[threadIdx.x + off];
            s_sq[threadIdx.x]  += s_sq[threadIdx.x + off];
        }
        __syncthreads();
    }
    float mean = s_sum[0] * (1.f / SPATIAL);
    float var  = s_sq[0] * (1.f / SPATIAL) - mean * mean;
    float rstd = rsqrtf(var + 1e-5f);
    for (int i = threadIdx.x; i < SPATIAL; i += 256) {
        float v = (src[i] - mean) * rstd;
        outp[(size_t)oc * SPATIAL + i] = 0.5f * v * (1.f + erff(v * 0.70710678118654752f));
    }
}

// ---------------------------------------------------------------------------
extern "C" void kernel_launch(void* const* d_in, const int* in_sizes, int n_in,
                              void* d_out, int out_size, void* d_ws, size_t ws_size,
                              hipStream_t stream) {
    const float* cam_feat = (const float*)d_in[0]; // (1,6,128,56,100)
    const float* intrins  = (const float*)d_in[1]; // (1,6,4,4)
    const float* rots     = (const float*)d_in[2]; // (1,6,3,3)
    const float* trans    = (const float*)d_in[3]; // (1,6,3)
    const float* conv_w   = (const float*)d_in[4]; // (128,1024,3,3)
    float* outp = (float*)d_out;                   // (1,128,128,128)

    // workspace layout (all 256B aligned)
    char* ws = (char*)d_ws;
    const size_t off_xf   = 0;                                   // 768 B -> 1024
    const size_t off_wp   = 1024;                                // 2,359,296 B
    const size_t off_bev  = off_wp + (size_t)9 * CDIM * ICDIM * 2;
    const size_t off_conv = off_bev + BEV_ELEMS * 2;             // bev: 34.6 MB
    float*          xf       = (float*)(ws + off_xf);
    unsigned short* wp       = (unsigned short*)(ws + off_wp);
    unsigned short* bev      = (unsigned short*)(ws + off_bev);
    float*          conv_out = (float*)(ws + off_conv);

    xform_kernel<<<1, 32, 0, stream>>>(intrins, rots, trans, xf);

    const long n4 = (long)(BEV_ELEMS / 8);   // uint4 = 8 bf16
    bev_zero_kernel<<<(unsigned)((n4 + 255) / 256), 256, 0, stream>>>((uint4*)bev, n4);

    const int nw = 9 * CDIM * ICDIM;
    wpack_kernel<<<(nw + 255) / 256, 256, 0, stream>>>(conv_w, wp, nw);

    sample_kernel<<<(ZDIM * YDIM * XDIM) / 8, 256, 0, stream>>>(cam_feat, xf, bev);

    conv_kernel<<<dim3(2, ZDIM), 128, 0, stream>>>(bev, wp, conv_out);

    norm_gelu_kernel<<<CDIM, 256, 0, stream>>>(conv_out, outp);
}
